// TypeAreaScaledDotProductAttention_36507222016297
// MI455X (gfx1250) — matrix operand order repaired and checked by reading, not compiled, not run on hardware
//
#include <hip/hip_runtime.h>

typedef __attribute__((ext_vector_type(16))) _Float16 v16h;
typedef __attribute__((ext_vector_type(8)))  _Float16 v8h;
typedef __attribute__((ext_vector_type(4)))  _Float16 v4h;
typedef __attribute__((ext_vector_type(8)))  float    v8f;

#define B_  2
#define H_  8
#define S_  4096
#define D_  64
#define KT_ (S_ / 64)
// fold 1/sqrt(64) and log2(e) into Q so softmax runs in exp2 domain
#define QSCALE (0.125f * 1.44269504088896340736f)

static __device__ __forceinline__ v8f wmma_f16(v16h a, v16h b, v8f c) {
    // D = A(16x32 f16) * B(32x16 f16) + C(16x16 f32)
    return __builtin_amdgcn_wmma_f32_16x16x32_f16(
        false, a, false, b, (short)0, c, false, false);
}

static __device__ __forceinline__ v16h ld_frag_lds(const _Float16* p0, const _Float16* p1) {
    // two ds_load_b128 concatenated into one v16h fragment
    v8h a = *(const v8h*)p0;
    v8h b = *(const v8h*)p1;
    return __builtin_shufflevector(a, b, 0,1,2,3,4,5,6,7,8,9,10,11,12,13,14,15);
}

static __device__ __forceinline__ v16h splat16(_Float16 x) {
    v16h r;
    #pragma unroll
    for (int i = 0; i < 16; ++i) r[i] = x;
    return r;
}

// ---- K/V tile staging: float4 global loads, fused add + f32->f16,
//      V rows of masked keys zeroed (mask folded off the critical path) ----
static __device__ __forceinline__ void stage_load(
    const float* __restrict__ ka, const float* __restrict__ ks,
    const float* __restrict__ va, const float* __restrict__ vs,
    const int* __restrict__ mask, size_t maskB,
    size_t baseBH, int kb, int tid,
    float4* kaR, float4* ksR, float4* vaR, float4* vsR, int* mkR)
{
    #pragma unroll
    for (int i = 0; i < 4; ++i) {
        const int idx = i * 256 + tid;          // 0..1023 quads
        const int key = idx >> 4;               // 0..63
        const int d4  = (idx & 15) * 4;         // 0..60
        const size_t g = baseBH + (size_t)(kb + key) * D_ + d4;
        kaR[i] = *(const float4*)&ka[g];
        ksR[i] = *(const float4*)&ks[g];
        vaR[i] = *(const float4*)&va[g];
        vsR[i] = *(const float4*)&vs[g];
        mkR[i] = mask[maskB + kb + key];
    }
}

static __device__ __forceinline__ void stage_store(
    _Float16 (*KtB)[72], _Float16 (*VsB)[72], int tid,
    const float4* kaR, const float4* ksR, const float4* vaR, const float4* vsR,
    const int* mkR)
{
    #pragma unroll
    for (int i = 0; i < 4; ++i) {
        const int idx = i * 256 + tid;
        const int key = idx >> 4;
        const int d4  = (idx & 15) * 4;
        const float kf[4] = { kaR[i].x + ksR[i].x, kaR[i].y + ksR[i].y,
                              kaR[i].z + ksR[i].z, kaR[i].w + ksR[i].w };
        #pragma unroll
        for (int j = 0; j < 4; ++j)
            KtB[key][d4 + j] = (_Float16)kf[j];   // row-major store
        const float mf = mkR[i] ? 1.0f : 0.0f;    // zero V row if masked
        v4h vh;
        vh[0] = (_Float16)((vaR[i].x + vsR[i].x) * mf);
        vh[1] = (_Float16)((vaR[i].y + vsR[i].y) * mf);
        vh[2] = (_Float16)((vaR[i].z + vsR[i].z) * mf);
        vh[3] = (_Float16)((vaR[i].w + vsR[i].w) * mf);
        #pragma unroll
        for (int j = 0; j < 4; ++j)
            VsB[d4 + j][key] = vh[j];             // transposed store
    }
}

__global__ __launch_bounds__(256)
void TypeAreaScaledDotProductAttention_36507222016297_kernel(
    const float* __restrict__ q_a, const float* __restrict__ k_a,
    const float* __restrict__ v_a, const float* __restrict__ q_s,
    const float* __restrict__ k_s, const float* __restrict__ v_s,
    const int*   __restrict__ mask, float* __restrict__ out)
{
    // double-buffered K/V tiles + per-wave P scratch (row stride 144B)
    __shared__ __align__(16) _Float16 Kt[2][64][72];   // [buf][key][d]
    __shared__ __align__(16) _Float16 Vs[2][64][72];   // [buf][d][key]
    __shared__ __align__(16) _Float16 Ps[8][16][72];   // [wave][row][key]

    const int tid  = threadIdx.x;
    const int lane = tid & 31;
    const int wave = tid >> 5;
    const int half = lane >> 4;
    const int col  = lane & 15;
    const int s8   = half * 8;

    const int qt = blockIdx.x;
    const int h  = blockIdx.y;
    const int b  = blockIdx.z;

    const size_t baseBH = ((size_t)(b * H_ + h)) * S_ * D_;
    const size_t maskB  = (size_t)b * S_;

    // ---- Q fragments (A-layout), prescaled into exp2 domain ----
    const int qrowA = qt * 128 + wave * 16 + col;
    const size_t qoff = baseBH + (size_t)qrowA * D_;
    v16h qlo, qhi;
    #pragma unroll
    for (int r = 0; r < 4; ++r) {
        const int d0 = r * 16 + s8;
        #pragma unroll
        for (int j = 0; j < 8; ++j) {
            float q = (q_a[qoff + d0 + j] + q_s[qoff + d0 + j]) * QSCALE;
            if (r < 2) qlo[r * 8 + j]       = (_Float16)q;
            else       qhi[(r - 2) * 8 + j] = (_Float16)q;
        }
    }

    v8f acc[4] = {{}, {}, {}, {}};
    v8f accL = {};               // masked row sums via WMMA
    float m_prev = -3.0e38f;     // wave-uniform running shift (exp2 domain)

    // ---- prologue: stage tile 0 ----
    {
        float4 kaR[4], ksR[4], vaR[4], vsR[4]; int mkR[4];
        stage_load(k_a, k_s, v_a, v_s, mask, maskB, baseBH, 0, tid,
                   kaR, ksR, vaR, vsR, mkR);
        stage_store(Kt[0], Vs[0], tid, kaR, ksR, vaR, vsR, mkR);
    }
    __syncthreads();

    for (int kt = 0; kt < KT_; ++kt) {
        const int cur = kt & 1;
        const int kb  = kt * 64;
        const bool more = (kt + 1) < KT_;

        // mask-weighted "ones" for the l-WMMA (B[key][n] = m(key) for all n)
        v16h mo0, mo1;
        #pragma unroll
        for (int s = 0; s < 16; ++s) {
            const int kk = (s < 8) ? (s8 + s) : (8 + s8 + s);
            mo0[s] = (_Float16)(mask[maskB + kb + kk] ? 1.0f : 0.0f);
            mo1[s] = (_Float16)(mask[maskB + kb + 32 + kk] ? 1.0f : 0.0f);
        }

        // issue next tile's global loads before the matrix work
        float4 kaR[4], ksR[4], vaR[4], vsR[4]; int mkR[4];
        if (more) {
            stage_load(k_a, k_s, v_a, v_s, mask, maskB, baseBH, kb + 64, tid,
                       kaR, ksR, vaR, vsR, mkR);
            if (kt + 2 < KT_) {  // warm L2/L1 two tiles ahead (global_prefetch_b8)
                const int key = tid >> 4;
                const int d4  = (tid & 15) * 4;
                const size_t g = baseBH + (size_t)(kb + 128 + key) * D_ + d4;
                __builtin_prefetch(&k_a[g], 0, 1);
                __builtin_prefetch(&k_s[g], 0, 1);
                __builtin_prefetch(&v_a[g], 0, 1);
                __builtin_prefetch(&v_s[g], 0, 1);
            }
        }

        // ---- scores: 8 WMMAs (no mask ops; softmax is shift-invariant) ----
        v8f sc[4];
        #pragma unroll
        for (int t = 0; t < 4; ++t) {
            v16h bk0 = ld_frag_lds(&Kt[cur][t * 16 + col][s8],      &Kt[cur][t * 16 + col][16 + s8]);
            v16h bk1 = ld_frag_lds(&Kt[cur][t * 16 + col][32 + s8], &Kt[cur][t * 16 + col][48 + s8]);
            v8f z = {};
            sc[t] = wmma_f16(qlo, bk0, z);
            sc[t] = wmma_f16(qhi, bk1, sc[t]);
        }

        // park the staged tile into the alternate LDS buffer (loads have had
        // the whole score phase to land)
        if (more)
            stage_store(Kt[cur ^ 1], Vs[cur ^ 1], tid, kaR, ksR, vaR, vsR, mkR);

        // ---- wave-uniform online shift: softmax only needs a row-uniform m
        // with  tile_max <= m  (p<=1 fits f16) and m within ~20 of each row's
        // max (f16 headroom); the wave-wide tile max satisfies both.
        float tm;
        {
            float a0 = fmaxf(fmaxf(sc[0][0], sc[0][1]), fmaxf(sc[0][2], sc[0][3]));
            float a1 = fmaxf(fmaxf(sc[0][4], sc[0][5]), fmaxf(sc[0][6], sc[0][7]));
            #pragma unroll
            for (int t = 1; t < 4; ++t) {
                a0 = fmaxf(a0, fmaxf(fmaxf(sc[t][0], sc[t][1]), fmaxf(sc[t][2], sc[t][3])));
                a1 = fmaxf(a1, fmaxf(fmaxf(sc[t][4], sc[t][5]), fmaxf(sc[t][6], sc[t][7])));
            }
            tm = fmaxf(a0, a1);
            tm = fmaxf(tm, __shfl_xor(tm, 1,  32));
            tm = fmaxf(tm, __shfl_xor(tm, 2,  32));
            tm = fmaxf(tm, __shfl_xor(tm, 4,  32));
            tm = fmaxf(tm, __shfl_xor(tm, 8,  32));
            tm = fmaxf(tm, __shfl_xor(tm, 16, 32));
        }
        const float mn    = fmaxf(m_prev, tm);
        const float scale = __builtin_amdgcn_exp2f(m_prev - mn);
        m_prev = mn;
        accL *= scale;
        #pragma unroll
        for (int t = 0; t < 4; ++t)
            acc[t] *= scale;
        #pragma unroll
        for (int t = 0; t < 4; ++t)
            #pragma unroll
            for (int v = 0; v < 8; ++v)
                sc[t][v] = __builtin_amdgcn_exp2f(sc[t][v] - mn);  // p in place, p<=1

        // ---- C-layout -> A-layout via per-wave LDS scratch (wave-local:
        // same-wave DS ops are in-order, no block barrier needed) ----
        #pragma unroll
        for (int t = 0; t < 4; ++t)
            #pragma unroll
            for (int v = 0; v < 8; ++v)
                Ps[wave][v + 8 * half][t * 16 + col] = (_Float16)sc[t][v];
        __builtin_amdgcn_wave_barrier();

        const v16h pa0 = ld_frag_lds(&Ps[wave][col][s8],      &Ps[wave][col][16 + s8]);
        const v16h pa1 = ld_frag_lds(&Ps[wave][col][32 + s8], &Ps[wave][col][48 + s8]);

        // ---- hoist all V fragments so the 10 WMMAs run without DS stalls ----
        v16h bv[8];
        #pragma unroll
        for (int t = 0; t < 4; ++t) {
            bv[2 * t]     = ld_frag_lds(&Vs[cur][t * 16 + col][s8],      &Vs[cur][t * 16 + col][16 + s8]);
            bv[2 * t + 1] = ld_frag_lds(&Vs[cur][t * 16 + col][32 + s8], &Vs[cur][t * 16 + col][48 + s8]);
        }
        #pragma unroll
        for (int t = 0; t < 4; ++t)
            acc[t] = wmma_f16(pa0, bv[2 * t], acc[t]);
        accL = wmma_f16(pa0, mo0, accL);   // l += P x mask (exact masked row sum)
        #pragma unroll
        for (int t = 0; t < 4; ++t)
            acc[t] = wmma_f16(pa1, bv[2 * t + 1], acc[t]);
        accL = wmma_f16(pa1, mo1, accL);

        __syncthreads();   // staged alt buffer ready; cur buffer free for reuse
    }

    // ---- epilogue ----
    float inv[8];
    #pragma unroll
    for (int v = 0; v < 8; ++v)
        inv[v] = __builtin_amdgcn_rcpf(accL[v]);

    const int qrow0 = qt * 128 + wave * 16 + 8 * half;
    #pragma unroll
    for (int t = 0; t < 4; ++t) {
        #pragma unroll
        for (int v = 0; v < 8; ++v) {
            const size_t o = baseBH + (size_t)(qrow0 + v) * D_ + t * 16 + col;
            out[o] = acc[t][v] * inv[v];
        }
    }
}

extern "C" void kernel_launch(void* const* d_in, const int* in_sizes, int n_in,
                              void* d_out, int out_size, void* d_ws, size_t ws_size,
                              hipStream_t stream) {
    const float* q_a = (const float*)d_in[0];
    const float* k_a = (const float*)d_in[1];
    const float* v_a = (const float*)d_in[2];
    const float* q_s = (const float*)d_in[3];
    const float* k_s = (const float*)d_in[4];
    const float* v_s = (const float*)d_in[5];
    const int*   msk = (const int*)d_in[6];
    float* out = (float*)d_out;

    dim3 grid(S_ / 128, H_, B_);   // 32 x 8 x 2
    dim3 block(256);               // 8 wave32
    TypeAreaScaledDotProductAttention_36507222016297_kernel<<<grid, block, 0, stream>>>(
        q_a, k_a, v_a, q_s, k_s, v_s, msk, out);
}
